// GCNInfer_18141941859039
// MI455X (gfx1250) — matrix operand order repaired and checked by reading, not compiled, not run on hardware
//
#include <hip/hip_runtime.h>

#define NN 100000   // nodes
#define NE 1600000  // edges
#define DF 128      // feature / hidden dim (all gather dims are 128)
#define NC 64       // classes

typedef __attribute__((ext_vector_type(8)))  __bf16 v8bf;
typedef __attribute__((ext_vector_type(16))) __bf16 v16bf;
typedef __attribute__((ext_vector_type(8)))  float  v8f;

__device__ __forceinline__ unsigned short f2bf(float f) {
    unsigned int u = __float_as_uint(f);
    u += 0x7FFFu + ((u >> 16) & 1u);   // round-to-nearest-even
    return (unsigned short)(u >> 16);
}
__device__ __forceinline__ float bf2f(unsigned int b) {
    return __uint_as_float(b << 16);
}

// fp32 -> bf16, 4 elements per thread (features prepass)
__global__ __launch_bounds__(256) void cvt_bf16_kernel(const float* __restrict__ in,
                                                       unsigned short* __restrict__ out, int n4) {
    int i = blockIdx.x * blockDim.x + threadIdx.x;
    if (i >= n4) return;
    float4 v = ((const float4*)in)[i];
    ushort4 o;
    o.x = f2bf(v.x); o.y = f2bf(v.y); o.z = f2bf(v.z); o.w = f2bf(v.w);
    ((ushort4*)out)[i] = o;
}

// W[K,N] fp32 -> Wt[N,K] bf16 (tiny, once per weight)
__global__ __launch_bounds__(256) void wtrans_kernel(const float* __restrict__ W,
                                                     unsigned short* __restrict__ Wt, int K, int N) {
    int i = blockIdx.x * blockDim.x + threadIdx.x;
    if (i >= K * N) return;
    int k = i / N, n = i % N;
    Wt[n * K + k] = f2bf(W[i]);
}

__global__ __launch_bounds__(256) void zero_kernel(float4* __restrict__ p, int n4) {
    int i = blockIdx.x * blockDim.x + threadIdx.x;
    if (i >= n4) return;
    p[i] = make_float4(0.f, 0.f, 0.f, 0.f);
}

// One wave per edge; lane handles 4 consecutive columns.
// Gather bf16 row (L2-resident: 25.6 MB << 192 MB L2), scatter-add fp32 with
// native f32 atomics (relaxed, agent scope -> global_atomic_add_f32, L2 RMW).
__global__ __launch_bounds__(256) void scatter_kernel(const unsigned short* __restrict__ Hb,
                                                      const int* __restrict__ src,
                                                      const int* __restrict__ dst,
                                                      float* __restrict__ agg) {
    int gw = (blockIdx.x * blockDim.x + threadIdx.x) >> 5;
    if (gw >= NE) return;
    int lane = threadIdx.x & 31;
    int s = src[gw];
    int d = dst[gw];
    uint2 v = *(const uint2*)(Hb + (size_t)s * DF + lane * 4);
    float f0 = bf2f(v.x & 0xFFFFu);
    float f1 = bf2f(v.x >> 16);
    float f2 = bf2f(v.y & 0xFFFFu);
    float f3 = bf2f(v.y >> 16);
    float* a = agg + (size_t)d * DF + lane * 4;
    __hip_atomic_fetch_add(a + 0, f0, __ATOMIC_RELAXED, __HIP_MEMORY_SCOPE_AGENT);
    __hip_atomic_fetch_add(a + 1, f1, __ATOMIC_RELAXED, __HIP_MEMORY_SCOPE_AGENT);
    __hip_atomic_fetch_add(a + 2, f2, __ATOMIC_RELAXED, __HIP_MEMORY_SCOPE_AGENT);
    __hip_atomic_fetch_add(a + 3, f3, __ATOMIC_RELAXED, __HIP_MEMORY_SCOPE_AGENT);
}

// Xb[n,c] = bf16(agg[n,c] * norm[n]); thread per (node, 4 cols)
__global__ __launch_bounds__(256) void scale_cvt_kernel(const float* __restrict__ agg,
                                                        const float* __restrict__ norm,
                                                        unsigned short* __restrict__ Xb) {
    int i = blockIdx.x * blockDim.x + threadIdx.x;
    if (i >= NN * 32) return;
    int node = i >> 5;
    int c = (i & 31) * 4;
    float nm = norm[node];
    float4 v = *(const float4*)(agg + (size_t)node * DF + c);
    ushort4 o;
    o.x = f2bf(v.x * nm); o.y = f2bf(v.y * nm);
    o.z = f2bf(v.z * nm); o.w = f2bf(v.w * nm);
    *(ushort4*)(Xb + (size_t)node * DF + c) = o;
}

// out[M,NOUT] = relu?(Xb[M,128] @ W + bias); W passed transposed as Wt[NOUT,128] bf16.
// One wave computes a 16x16 tile via 4x v_wmma_f32_16x16x32_bf16 (K=128).
// Fragment layouts used here (contiguous k, shared by A and B) (wave32, h = lane>>4, r = lane&15):
//   A (16x32, contiguous):     row M = r; elements 0..7  -> K = s*32 + h*16 + {0..7}
//                                        elements 8..15 -> K = s*32 + h*16 + 8 + {0..7}
//   B (32x16, same k order as A):
//       col N = r; elements 0..15 -> K = s*32 + h*16 + {0..15} (contiguous 32B load)
//   C/D: element v -> row M = h*8 + v, col N = r
template <int NOUT, bool RELU, bool OUTBF>
__global__ __launch_bounds__(256) void gcn_gemm_kernel(const unsigned short* __restrict__ Xb,
                                                       const unsigned short* __restrict__ Wt,
                                                       const float* __restrict__ bias,
                                                       unsigned short* __restrict__ out_bf,
                                                       float* __restrict__ out_f) {
    const int TN = NOUT / 16;
    int wid = (blockIdx.x * blockDim.x + threadIdx.x) >> 5;   // wave-uniform
    if (wid >= (NN / 16) * TN) return;                        // whole wave exits together
    int lane = threadIdx.x & 31;
    int r = lane & 15;
    int h = lane >> 4;
    int mt = wid / TN, nt = wid % TN;
    int m0 = mt * 16, n0 = nt * 16;

    const __bf16* arow = (const __bf16*)Xb + (size_t)(m0 + r) * DF + h * 16;
    const __bf16* brow = (const __bf16*)Wt + (size_t)(n0 + r) * DF + h * 16;

    float bv = bias[n0 + r];        // D element v lives at col n0+r for all 8 accum regs
    v8f acc;
#pragma unroll
    for (int i = 0; i < 8; ++i) acc[i] = bv;

#pragma unroll
    for (int s = 0; s < 4; ++s) {
        v8bf alo = *(const v8bf*)(arow + s * 32);
        v8bf ahi = *(const v8bf*)(arow + s * 32 + 8);
        v16bf A = __builtin_shufflevector(alo, ahi, 0,1,2,3,4,5,6,7,8,9,10,11,12,13,14,15);
        v16bf B = *(const v16bf*)(brow + s * 32);   // contiguous K chunk of 16
        acc = __builtin_amdgcn_wmma_f32_16x16x32_bf16(false, A, false, B, (short)0, acc,
                                                      false, false);
    }

#pragma unroll
    for (int v = 0; v < 8; ++v) {
        float x = acc[v];
        if (RELU) x = fmaxf(x, 0.0f);
        size_t idx = (size_t)(m0 + h * 8 + v) * NOUT + (n0 + r);
        if (OUTBF) out_bf[idx] = f2bf(x);
        else       out_f[idx] = x;
    }
}

extern "C" void kernel_launch(void* const* d_in, const int* in_sizes, int n_in,
                              void* d_out, int out_size, void* d_ws, size_t ws_size,
                              hipStream_t stream) {
    const float* features = (const float*)d_in[0];
    const float* norm     = (const float*)d_in[1];
    const float* W0       = (const float*)d_in[2];
    const float* b0       = (const float*)d_in[3];
    const float* W1       = (const float*)d_in[4];
    const float* b1       = (const float*)d_in[5];
    const float* W2       = (const float*)d_in[6];
    const float* b2       = (const float*)d_in[7];
    const int*   src      = (const int*)d_in[8];
    const int*   dst      = (const int*)d_in[9];
    float* out = (float*)d_out;

    char* ws = (char*)d_ws;
    size_t off = 0;
    auto walloc = [&](size_t bytes) -> void* {
        off = (off + 255) & ~(size_t)255;
        void* p = ws + off;
        off += bytes;
        return p;
    };
    unsigned short* Hb  = (unsigned short*)walloc((size_t)NN * DF * 2); // node feats, bf16
    unsigned short* Xb  = (unsigned short*)walloc((size_t)NN * DF * 2); // scaled agg, bf16
    float*          agg = (float*)walloc((size_t)NN * DF * 4);          // fp32 accumulators
    unsigned short* W0t = (unsigned short*)walloc((size_t)DF * DF * 2);
    unsigned short* W1t = (unsigned short*)walloc((size_t)DF * DF * 2);
    unsigned short* W2t = (unsigned short*)walloc((size_t)DF * NC * 2);

    const int B = 256;
    const int n4 = NN * DF / 4;                 // 3.2M float4
    const int zgrid = (n4 + B - 1) / B;
    const int scgrid = (NN * 32 + B - 1) / B;
    const int egrid = (NE * 32) / B;            // wave per edge: 200000 blocks

    // prepasses: bf16 features + bf16 transposed weights
    cvt_bf16_kernel<<<zgrid, B, 0, stream>>>(features, Hb, n4);
    wtrans_kernel<<<(DF * DF + B - 1) / B, B, 0, stream>>>(W0, W0t, DF, DF);
    wtrans_kernel<<<(DF * DF + B - 1) / B, B, 0, stream>>>(W1, W1t, DF, DF);
    wtrans_kernel<<<(DF * NC + B - 1) / B, B, 0, stream>>>(W2, W2t, DF, NC);

    const int waves128 = (NN / 16) * (DF / 16); // 50000
    const int waves64  = (NN / 16) * (NC / 16); // 25000

    // ---- layer 0 ----
    zero_kernel<<<zgrid, B, 0, stream>>>((float4*)agg, n4);
    scatter_kernel<<<egrid, B, 0, stream>>>(Hb, src, dst, agg);
    scale_cvt_kernel<<<scgrid, B, 0, stream>>>(agg, norm, Xb);
    gcn_gemm_kernel<DF, true, true><<<(waves128 * 32) / B, B, 0, stream>>>(Xb, W0t, b0, Hb, nullptr);

    // ---- layer 1 ----
    zero_kernel<<<zgrid, B, 0, stream>>>((float4*)agg, n4);
    scatter_kernel<<<egrid, B, 0, stream>>>(Hb, src, dst, agg);
    scale_cvt_kernel<<<scgrid, B, 0, stream>>>(agg, norm, Xb);
    gcn_gemm_kernel<DF, true, true><<<(waves128 * 32) / B, B, 0, stream>>>(Xb, W1t, b1, Hb, nullptr);

    // ---- layer 2 ----
    zero_kernel<<<zgrid, B, 0, stream>>>((float4*)agg, n4);
    scatter_kernel<<<egrid, B, 0, stream>>>(Hb, src, dst, agg);
    scale_cvt_kernel<<<scgrid, B, 0, stream>>>(agg, norm, Xb);
    gcn_gemm_kernel<NC, false, false><<<(waves64 * 32) / B, B, 0, stream>>>(Xb, W2t, b2, nullptr, out);
}
